// DeltaFiberUpdate_73933567033934
// MI455X (gfx1250) — matrix operand order repaired and checked by reading, not compiled, not run on hardware
//
#include <hip/hip_runtime.h>
#include <hip/hip_bf16.h>
#include <math.h>

#define DEV __device__ __forceinline__

typedef __attribute__((ext_vector_type(16))) __bf16 v16bf;
typedef __attribute__((ext_vector_type(8)))  __bf16 v8bf;
typedef __attribute__((ext_vector_type(4)))  __bf16 v4bf;
typedef __attribute__((ext_vector_type(8)))  float  v8f;

constexpr int Bc = 4, Tc = 2048, Pc = 16, Jc = 256, Nc = 64, Hc = 4, MEMc = 64, Kc = 192;
constexpr float LN_EPS = 1e-5f;
constexpr int WROWS = 208;  // 64 q + 64 k + 64 v + 4 beta + 4 alpha + 8 pad

constexpr size_t NTOK    = (size_t)Nc * Tc;       // 131072 tokens
constexpr size_t X_ELEMS = NTOK * Jc;             // 33,554,432

// workspace layout (bytes), all regions 256B aligned
constexpr size_t OFF_XBF   = 0;
constexpr size_t SZ_XBF    = X_ELEMS * 2;
constexpr size_t OFF_WCAT  = OFF_XBF + SZ_XBF;
constexpr size_t SZ_WCAT   = (size_t)WROWS * Jc * 2;
constexpr size_t OFF_WOUT  = OFF_WCAT + SZ_WCAT;
constexpr size_t SZ_WOUT   = (size_t)Kc * MEMc * 2;
constexpr size_t OFF_Q     = OFF_WOUT + SZ_WOUT;
constexpr size_t SZ_QKV    = NTOK * MEMc * 4;
constexpr size_t OFF_K     = OFF_Q + SZ_QKV;
constexpr size_t OFF_V     = OFF_K + SZ_QKV;
constexpr size_t OFF_BETA  = OFF_V + SZ_QKV;
constexpr size_t SZ_GATE   = NTOK * Hc * 4;
constexpr size_t OFF_ALPHA = OFF_BETA + SZ_GATE;
constexpr size_t OFF_O     = OFF_ALPHA + SZ_GATE;

DEV int lane_id() { return (int)(threadIdx.x & 31); }

DEV float shflf(float v, int srcLane) {
  return __int_as_float(__builtin_amdgcn_ds_bpermute(srcLane << 2, __float_as_int(v)));
}

DEV float sigmoidf(float x) { return 1.f / (1.f + __expf(-x)); }

// pairwise tree dot of two 16-element register arrays: depth 4 (+1 mul)
DEV float dot16(const float* a, const float* b) {
  float p[16];
#pragma unroll
  for (int j = 0; j < 16; ++j) p[j] = a[j] * b[j];
#pragma unroll
  for (int s = 8; s >= 1; s >>= 1) {
#pragma unroll
    for (int j = 0; j < s; ++j) p[j] += p[j + s];
  }
  return p[0];
}

// -------- kernel 1: fp32 -> bf16 copy of x --------
__global__ void pack_x_kernel(const float* __restrict__ x, __bf16* __restrict__ xb) {
  size_t i = ((size_t)blockIdx.x * blockDim.x + threadIdx.x) * 4;
  if (i + 3 < X_ELEMS) {
    float4 v = *(const float4*)(x + i);
    v4bf o;
    o[0] = (__bf16)v.x; o[1] = (__bf16)v.y; o[2] = (__bf16)v.z; o[3] = (__bf16)v.w;
    *(v4bf*)(xb + i) = o;
  }
}

// -------- kernel 2: pack weights to bf16 --------
__global__ void pack_w_kernel(const float* __restrict__ Wq, const float* __restrict__ Wk,
                              const float* __restrict__ Wv, const float* __restrict__ Wbw,
                              const float* __restrict__ Waw, const float* __restrict__ Wout,
                              __bf16* __restrict__ wcat, __bf16* __restrict__ wob) {
  int idx = blockIdx.x * blockDim.x + threadIdx.x;
  const int total1 = WROWS * Jc;  // 53248
  if (idx < total1) {
    int r = idx / Jc, c = idx - r * Jc;
    float v = 0.f;
    if (r < 64)       v = Wq[r * Jc + c];
    else if (r < 128) v = Wk[(r - 64) * Jc + c];
    else if (r < 192) v = Wv[(r - 128) * Jc + c];
    else if (r < 196) v = Wbw[(r - 192) * Jc + c];
    else if (r < 200) v = Waw[(r - 196) * Jc + c];
    wcat[idx] = (__bf16)v;
  } else {
    int j = idx - total1;
    if (j < Kc * MEMc) wob[j] = (__bf16)Wout[j];
  }
}

// -------- kernel 3: QKV + gate projection, fused LN / k-norm / sigmoids --------
__global__ void __launch_bounds__(128) proj_kernel(
    const __bf16* __restrict__ xb, const __bf16* __restrict__ wcat,
    float* __restrict__ Q, float* __restrict__ Kd, float* __restrict__ V,
    float* __restrict__ BETA, float* __restrict__ ALPHA,
    const float* __restrict__ curvature, const float* __restrict__ entropy,
    const float* __restrict__ wbeta_b, const float* __restrict__ walpha_b,
    const float* __restrict__ curv_w, const float* __restrict__ ent_w,
    const float* __restrict__ lnqw, const float* __restrict__ lnqb,
    const float* __restrict__ lnkw, const float* __restrict__ lnkb) {
  const int lane = lane_id();
  const int wave = threadIdx.x >> 5;
  const int tile = blockIdx.x * 4 + wave;        // 8192 tiles of 16 tokens
  const int n    = tile >> 7;                     // Tc/16 == 128
  const int t0   = (tile & 127) << 4;
  const int col  = lane & 15;
  const int half = lane >> 4;

  // A fragments: 16 tokens x 256, bf16 layout (row = lane&15, K-halves by lane>>4)
  const __bf16* arow = xb + ((size_t)n * Tc + t0 + col) * Jc;
  v16bf afrag[8];
#pragma unroll
  for (int kk = 0; kk < 8; ++kk) {
    const int kb = kk * 32;
    v8bf lo = *(const v8bf*)(arow + kb + 16 * half);
    v8bf hi = *(const v8bf*)(arow + kb + 16 * half + 8);
#pragma unroll
    for (int i = 0; i < 8; ++i) { afrag[kk][i] = lo[i]; afrag[kk][i + 8] = hi[i]; }
  }

  for (int nt = 0; nt < 13; ++nt) {
    const __bf16* brow = wcat + ((size_t)(nt * 16 + col)) * Jc + 16 * half;
    // preload all 8 B fragments into distinct registers so the 8 WMMAs can
    // issue back-to-back with staggered loadcnt waits instead of stalling
    v16bf bfrag[8];
#pragma unroll
    for (int kk = 0; kk < 8; ++kk) bfrag[kk] = *(const v16bf*)(brow + kk * 32);

    v8f acc = {};
#pragma unroll
    for (int kk = 0; kk < 8; ++kk) {
      acc = __builtin_amdgcn_wmma_f32_16x16x32_bf16(false, afrag[kk], false, bfrag[kk],
                                                    (short)0, acc, false, false);
    }

    if (nt < 12) {
      const bool isQ = (nt < 4);
      const bool isK = (nt >= 4) && (nt < 8);
      const int h = nt & 3;
      float lw = 1.f, lb = 0.f;
      if (isQ) { lw = lnqw[col]; lb = lnqb[col]; }
      else if (isK) { lw = lnkw[col]; lb = lnkb[col]; }
      float* dst = isQ ? Q : (isK ? Kd : V);
#pragma unroll
      for (int r = 0; r < 8; ++r) {
        const int t = t0 + r + 8 * half;
        float val = acc[r];
        float y;
        if (nt < 8) {
          // LayerNorm over the 16 columns (live across 16 lanes of this half)
          float s = val, s2 = val * val;
#pragma unroll
          for (int mm = 1; mm < 16; mm <<= 1) {
            s += shflf(s, lane ^ mm);
            s2 += shflf(s2, lane ^ mm);
          }
          const float mean = s * 0.0625f;
          const float var  = s2 * 0.0625f - mean * mean;
          y = (val - mean) * rsqrtf(var + LN_EPS) * lw + lb;
          if (isK) {
            float ss = y * y;
#pragma unroll
            for (int mm = 1; mm < 16; mm <<= 1) ss += shflf(ss, lane ^ mm);
            y = y / fmaxf(sqrtf(ss), 1e-12f);
          }
        } else {
          y = val;
        }
        dst[((size_t)n * Tc + t) * MEMc + h * 16 + col] = y;
      }
    } else {
      // gate tile: cols 0-3 beta, 4-7 alpha
      const int bidx = n >> 4;  // n // P
      if (col < 4) {
        const int h = col;
        const float bb = wbeta_b[h];
        const float cw = curv_w[h];
        const float Kv = fminf(fabsf(curvature[bidx]), 10.f);
#pragma unroll
        for (int r = 0; r < 8; ++r) {
          const int t = t0 + r + 8 * half;
          const float b1 = sigmoidf(acc[r] + bb);
          BETA[((size_t)n * Tc + t) * Hc + h] = sigmoidf(b1 + Kv * cw);
        }
      } else if (col < 8) {
        const int h = col - 4;
        const float ab = walpha_b[h];
        const float ew = ent_w[h];
        const float Sv = fminf(fmaxf(entropy[bidx], 0.f), 5.f);
#pragma unroll
        for (int r = 0; r < 8; ++r) {
          const int t = t0 + r + 8 * half;
          const float a1 = sigmoidf(acc[r] + ab);
          ALPHA[((size_t)n * Tc + t) * Hc + h] = sigmoidf(a1 + Sv * ew);
        }
      }
    }
  }
}

// -------- kernel 4: sequential delta-rule scan; 2 (n,h) states per wave --------
// Critical path per step is only Mk(tree) -> coef -> m update; o is decoupled via
// o = M_old q + coef * (k.q). Loads for t+1 are software-pipelined.
__global__ void __launch_bounds__(128) scan_kernel(
    const float* __restrict__ Q, const float* __restrict__ Kd, const float* __restrict__ V,
    const float* __restrict__ BETA, const float* __restrict__ ALPHA,
    float* __restrict__ O) {
  const int lane  = lane_id();
  const int half  = lane >> 4;
  const int row   = lane & 15;
  const int gwave = blockIdx.x * (blockDim.x >> 5) + (threadIdx.x >> 5);
  const int s = gwave * 2 + half;  // state id 0..255
  const int n = s >> 2;
  const int h = s & 3;
  const int base = lane & 16;      // shuffle base for this half

  const size_t vecBase  = ((size_t)n * Tc) * MEMc + h * 16 + row;
  const size_t gateBase = (size_t)n * Tc * Hc + h;

  float m[16];
#pragma unroll
  for (int j = 0; j < 16; ++j) m[j] = 0.f;

  // prologue: load step 0
  float kc = Kd[vecBase];
  float qc = Q[vecBase];
  float vc = V[vecBase];
  float bt = BETA[gateBase];
  float at = ALPHA[gateBase];

  for (int t = 0; t < Tc; ++t) {
    // issue next step's loads early (overlap with this step's VALU work)
    float kn = 0.f, qn = 0.f, vn = 0.f, bn = 0.f, an = 0.f;
    if (t + 1 < Tc) {
      const size_t vj = vecBase + (size_t)(t + 1) * MEMc;
      kn = Kd[vj]; qn = Q[vj]; vn = V[vj];
      bn = BETA[gateBase + (size_t)(t + 1) * Hc];
      an = ALPHA[gateBase + (size_t)(t + 1) * Hc];
    }
    if (t + 16 < Tc) {
      const size_t vp = vecBase + (size_t)(t + 16) * MEMc;
      __builtin_prefetch(&Kd[vp], 0, 1);
      __builtin_prefetch(&Q[vp], 0, 1);
      __builtin_prefetch(&V[vp], 0, 1);
    }

    float kj[16], qj[16];
#pragma unroll
    for (int j = 0; j < 16; ++j) kj[j] = shflf(kc, base + j);
#pragma unroll
    for (int j = 0; j < 16; ++j) qj[j] = shflf(qc, base + j);

    const float Mk = dot16(m, kj);   // (M k)[row]
    const float Mq = dot16(m, qj);   // (M q)[row]  (old M)
    const float kq = dot16(kj, qj);  // k . q (scalar, same in all lanes of half)

    const float coef = fmaf(bt, vc, -(at * Mk));  // -a*Mk[i] + b*v[i]
    O[vecBase + (size_t)t * MEMc] = fmaf(coef, kq, Mq);
#pragma unroll
    for (int j = 0; j < 16; ++j) m[j] = fmaf(coef, kj[j], m[j]);

    kc = kn; qc = qn; vc = vn; bt = bn; at = an;
  }
}

// -------- kernel 5: output projection 64 -> 192 (bf16 WMMA) + transpose store --------
__global__ void __launch_bounds__(128) outproj_kernel(
    const float* __restrict__ O, const __bf16* __restrict__ wob,
    const float* __restrict__ out_b, float* __restrict__ out) {
  const int lane = lane_id();
  const int wave = threadIdx.x >> 5;
  const int tile = blockIdx.x * 4 + wave;
  const int n    = tile >> 7;
  const int t0   = (tile & 127) << 4;
  const int col  = lane & 15;
  const int half = lane >> 4;
  const int b = n >> 4, p = n & 15;

  const float* arow = O + ((size_t)n * Tc + t0 + col) * MEMc;
  v16bf afrag[2];
#pragma unroll
  for (int kk = 0; kk < 2; ++kk) {
    const int kb = kk * 32;
    const float* lo = arow + kb + 16 * half;
    const float* hi = arow + kb + 16 * half + 8;
#pragma unroll
    for (int i = 0; i < 8; ++i) {
      afrag[kk][i]     = (__bf16)lo[i];
      afrag[kk][i + 8] = (__bf16)hi[i];
    }
  }

  for (int nt = 0; nt < 12; ++nt) {
    const __bf16* brow = wob + ((size_t)(nt * 16 + col)) * MEMc + 16 * half;
    v16bf bfrag[2];
#pragma unroll
    for (int kk = 0; kk < 2; ++kk) bfrag[kk] = *(const v16bf*)(brow + kk * 32);

    v8f acc = {};
#pragma unroll
    for (int kk = 0; kk < 2; ++kk) {
      acc = __builtin_amdgcn_wmma_f32_16x16x32_bf16(false, afrag[kk], false, bfrag[kk],
                                                    (short)0, acc, false, false);
    }
    const float bias = out_b[nt * 16 + col];
#pragma unroll
    for (int r = 0; r < 8; ++r) {
      const int t = t0 + r + 8 * half;
      out[(((size_t)b * Tc + t) * Pc + p) * Kc + nt * 16 + col] = acc[r] + bias;
    }
  }
}

extern "C" void kernel_launch(void* const* d_in, const int* in_sizes, int n_in,
                              void* d_out, int out_size, void* d_ws, size_t ws_size,
                              hipStream_t stream) {
  (void)in_sizes; (void)n_in; (void)out_size; (void)ws_size;
  const float* jr    = (const float*)d_in[0];
  const float* curv  = (const float*)d_in[1];
  const float* ent   = (const float*)d_in[2];
  const float* Wq    = (const float*)d_in[3];
  const float* Wk    = (const float*)d_in[4];
  const float* Wv    = (const float*)d_in[5];
  const float* Wbw   = (const float*)d_in[6];
  const float* Wbb   = (const float*)d_in[7];
  const float* Waw   = (const float*)d_in[8];
  const float* Wab   = (const float*)d_in[9];
  const float* curvw = (const float*)d_in[10];
  const float* entw  = (const float*)d_in[11];
  const float* Wout  = (const float*)d_in[12];
  const float* outb  = (const float*)d_in[13];
  const float* lnqw  = (const float*)d_in[14];
  const float* lnqb  = (const float*)d_in[15];
  const float* lnkw  = (const float*)d_in[16];
  const float* lnkb  = (const float*)d_in[17];
  float* out = (float*)d_out;
  char* ws = (char*)d_ws;

  __bf16* xb   = (__bf16*)(ws + OFF_XBF);
  __bf16* wcat = (__bf16*)(ws + OFF_WCAT);
  __bf16* wob  = (__bf16*)(ws + OFF_WOUT);
  float* Qb = (float*)(ws + OFF_Q);
  float* Kb = (float*)(ws + OFF_K);
  float* Vb = (float*)(ws + OFF_V);
  float* Bb = (float*)(ws + OFF_BETA);
  float* Ab = (float*)(ws + OFF_ALPHA);
  float* Ob = (float*)(ws + OFF_O);

  pack_x_kernel<<<(int)(X_ELEMS / 4 / 256), 256, 0, stream>>>(jr, xb);
  pack_w_kernel<<<(WROWS * Jc + Kc * MEMc + 255) / 256, 256, 0, stream>>>(
      Wq, Wk, Wv, Wbw, Waw, Wout, wcat, wob);
  proj_kernel<<<2048, 128, 0, stream>>>(xb, wcat, Qb, Kb, Vb, Bb, Ab,
                                        curv, ent, Wbb, Wab, curvw, entw,
                                        lnqw, lnqb, lnkw, lnkb);
  scan_kernel<<<32, 128, 0, stream>>>(Qb, Kb, Vb, Bb, Ab, Ob);
  outproj_kernel<<<2048, 128, 0, stream>>>(Ob, wob, outb, out);
}
